// FlashDiffusionModule_25872882991444
// MI455X (gfx1250) — matrix operand order repaired and checked by reading, not compiled, not run on hardware
//
#include <hip/hip_runtime.h>
#include <hip/hip_bf16.h>
#include <math.h>

#define ALPHA_C 1.0f
#define BETA_C  0.05f

typedef __attribute__((ext_vector_type(16))) _Float16 v16h;
typedef __attribute__((ext_vector_type(8)))  _Float16 v8h;
typedef __attribute__((ext_vector_type(8)))  float    v8f;

constexpr int DIM     = 128;              // feature dim (D == d == 128)
constexpr int WAVES   = 4;                // waves per block
constexpr int BLOCK   = WAVES * 32;       // 128 threads
constexpr int MT      = 2;                // 16-row A tiles per wave
constexpr int BM_WAVE = 16 * MT;          // 32 query rows per wave
constexpr int BM      = WAVES * BM_WAVE;  // 128 query rows per block
constexpr int BN      = 64;               // KV columns per main-loop step
constexpr int NT      = BN / 16;          // 4 column tiles per step
constexpr int SPLITS  = 4;                // KV splits (flash-decoding style)

// ---------------- prep kernels ----------------

__global__ void cvt_f16_kernel(const float* __restrict__ src,
                               _Float16* __restrict__ dst, int n) {
    int i = blockIdx.x * blockDim.x + threadIdx.x;
    if (i < n) dst[i] = (_Float16)src[i];
}

// Vt[d][m] = (f16) V[m][d]   (coalesced writes)
__global__ void transpose_v_kernel(const float* __restrict__ V,
                                   _Float16* __restrict__ Vt, int M) {
    int i = blockIdx.x * blockDim.x + threadIdx.x;
    if (i < M * DIM) {
        int d = i / M;
        int m = i - d * M;
        Vt[i] = (_Float16)V[(size_t)m * DIM + d];
    }
}

// a[i] = -beta*||q_i||^2 - alpha*log_d_q[i] + log h[i]
__global__ void row_bias_kernel(const float* __restrict__ Q,
                                const float* __restrict__ log_d_q,
                                const float* __restrict__ h,
                                float* __restrict__ a, int N) {
    int i = blockIdx.x * blockDim.x + threadIdx.x;
    if (i < N) {
        float s = 0.f;
        const float* q = Q + (size_t)i * DIM;
        #pragma unroll 8
        for (int k = 0; k < DIM; ++k) s += q[k] * q[k];
        a[i] = -BETA_C * s - ALPHA_C * log_d_q[i] + logf(h[i]);
    }
}

// b[j] = -beta*||k_j||^2 - alpha*log_d_k[j] - log h[j]
__global__ void col_bias_kernel(const float* __restrict__ K,
                                const float* __restrict__ log_d_k,
                                const float* __restrict__ h,
                                float* __restrict__ b, int M) {
    int j = blockIdx.x * blockDim.x + threadIdx.x;
    if (j < M) {
        float s = 0.f;
        const float* k = K + (size_t)j * DIM;
        #pragma unroll 8
        for (int kk = 0; kk < DIM; ++kk) s += k[kk] * k[kk];
        b[j] = -BETA_C * s - ALPHA_C * log_d_k[j] - logf(h[j]);
    }
}

// ---------------- main flash kernel (one KV split per blockIdx.y) ----------------

__device__ __forceinline__ v16h cat8(v8h lo, v8h hi) {
    return __builtin_shufflevector(lo, hi, 0,1,2,3,4,5,6,7,8,9,10,11,12,13,14,15);
}

__global__ __launch_bounds__(BLOCK, 1)
void flash_doob_kernel(const _Float16* __restrict__ Qh,
                       const _Float16* __restrict__ Kh,
                       const _Float16* __restrict__ Vt,
                       const float* __restrict__ abias,
                       const float* __restrict__ bbias,
                       float* __restrict__ Opart,   // [SPLITS][N][DIM]
                       float* __restrict__ mpart,   // [SPLITS][N]
                       float* __restrict__ lpart,   // [SPLITS][N]
                       int N, int M) {
    __shared__ _Float16 ldsP[WAVES][MT][16][BN];   // 16 KB: per-wave P staging

    const int tid  = threadIdx.x;
    const int wave = tid >> 5;
    const int lane = tid & 31;
    const int half = lane >> 4;      // 0 or 1
    const int l16  = lane & 15;

    const int rowBase = blockIdx.x * BM + wave * BM_WAVE;
    const int split   = blockIdx.y;
    const int kvLen   = M / SPLITS;
    const int jBeg    = split * kvLen;
    const int jEnd    = jBeg + kvLen;

    // ---- Q A-fragments for MT row-tiles, kept in registers all loop ----
    // A layout: row = lane&15; k in [kb, kb+8) U [kb+16, kb+24), kb = 8*(lane>=16)
    v16h qfrag[MT][4];
    #pragma unroll
    for (int u = 0; u < MT; ++u) {
        const _Float16* qp = Qh + (size_t)(rowBase + u * 16 + l16) * DIM + half * 8;
        #pragma unroll
        for (int ks = 0; ks < 4; ++ks) {
            v8h lo = *(const v8h*)(qp + ks * 32);
            v8h hi = *(const v8h*)(qp + ks * 32 + 16);
            qfrag[u][ks] = cat8(lo, hi);
        }
    }

    // per-row additive bias; C-layout row = v + 8*half (per row-tile)
    float a_v[MT][8];
    #pragma unroll
    for (int u = 0; u < MT; ++u)
        #pragma unroll
        for (int v = 0; v < 8; ++v)
            a_v[u][v] = abias[rowBase + u * 16 + v + 8 * half];

    v8f Ofrag[MT][8];
    #pragma unroll
    for (int u = 0; u < MT; ++u)
        #pragma unroll
        for (int t = 0; t < 8; ++t) Ofrag[u][t] = (v8f){};
    float mrow[MT][8], lrow[MT][8];
    #pragma unroll
    for (int u = 0; u < MT; ++u)
        #pragma unroll
        for (int v = 0; v < 8; ++v) { mrow[u][v] = -INFINITY; lrow[u][v] = 0.f; }

    for (int j0 = jBeg; j0 < jEnd; j0 += BN) {
        // ---- S(32x64) = Q(32x128) * K^T : each B fragment feeds MT WMMAs ----
        v8f acc[MT][NT];
        #pragma unroll
        for (int u = 0; u < MT; ++u)
            #pragma unroll
            for (int n = 0; n < NT; ++n) acc[u][n] = (v8f){};
        // B layout: col = lane&15; k in [kb, kb+8) U [kb+16, kb+24), kb = 8*(lane>=16)
        const _Float16* kp = Kh + (size_t)(j0 + l16) * DIM + half * 8;
        #pragma unroll
        for (int ks = 0; ks < 4; ++ks) {
            #pragma unroll
            for (int n = 0; n < NT; ++n) {
                const _Float16* kpn = kp + (size_t)(n * 16) * DIM + ks * 32;
                v16h b = cat8(*(const v8h*)(kpn), *(const v8h*)(kpn + 16));
                #pragma unroll
                for (int u = 0; u < MT; ++u)
                    acc[u][n] = __builtin_amdgcn_wmma_f32_16x16x32_f16(
                        false, qfrag[u][ks], false, b, (short)0, acc[u][n], false, false);
            }
        }

        float bc[NT];
        #pragma unroll
        for (int n = 0; n < NT; ++n) bc[n] = bbias[j0 + n * 16 + l16];

        // ---- online softmax over this 64-column slab, both row tiles ----
        float scale_v[MT][8];
        #pragma unroll
        for (int u = 0; u < MT; ++u) {
            #pragma unroll
            for (int v = 0; v < 8; ++v) {
                float s[NT];
                #pragma unroll
                for (int n = 0; n < NT; ++n)
                    s[n] = 2.f * BETA_C * acc[u][n][v] + a_v[u][v] + bc[n];
                float t = fmaxf(fmaxf(s[0], s[1]), fmaxf(s[2], s[3]));
                #pragma unroll
                for (int xm = 8; xm >= 1; xm >>= 1) t = fmaxf(t, __shfl_xor(t, xm, 32));
                float mnew = fmaxf(mrow[u][v], t);
                float p[NT], ps = 0.f;
                #pragma unroll
                for (int n = 0; n < NT; ++n) { p[n] = __expf(s[n] - mnew); ps += p[n]; }
                #pragma unroll
                for (int xm = 8; xm >= 1; xm >>= 1) ps += __shfl_xor(ps, xm, 32);
                float sc = __expf(mrow[u][v] - mnew);
                lrow[u][v] = lrow[u][v] * sc + ps;
                mrow[u][v] = mnew;
                scale_v[u][v] = sc;
                const int r = v + 8 * half;
                #pragma unroll
                for (int n = 0; n < NT; ++n)
                    ldsP[wave][u][r][n * 16 + l16] = (_Float16)p[n];
            }
        }
        // wave-private LDS: only need DS-counter ordering, not a barrier
        asm volatile("s_wait_dscnt 0" ::: "memory");

        // rescale running output
        #pragma unroll
        for (int u = 0; u < MT; ++u)
            #pragma unroll
            for (int t = 0; t < 8; ++t)
                #pragma unroll
                for (int v = 0; v < 8; ++v) Ofrag[u][t][v] *= scale_v[u][v];

        // P A-fragments from LDS (layout transform C -> A), k = 0..31 / 32..63
        v16h pfrag[MT][2];
        #pragma unroll
        for (int u = 0; u < MT; ++u)
            #pragma unroll
            for (int f = 0; f < 2; ++f) {
                const _Float16* pp = &ldsP[wave][u][l16][0] + f * 32 + half * 8;
                pfrag[u][f] = cat8(*(const v8h*)(pp), *(const v8h*)(pp + 16));
            }

        // ---- O(32x128) += P(32x64) * V(64x128); each Vt fragment feeds MT WMMAs ----
        const _Float16* vp = Vt + (size_t)l16 * M + j0 + half * 8;
        #pragma unroll
        for (int t = 0; t < 8; ++t) {
            const _Float16* vpt = vp + (size_t)(t * 16) * M;
            #pragma unroll
            for (int f = 0; f < 2; ++f) {
                v16h bf = cat8(*(const v8h*)(vpt + f * 32), *(const v8h*)(vpt + f * 32 + 16));
                #pragma unroll
                for (int u = 0; u < MT; ++u)
                    Ofrag[u][t] = __builtin_amdgcn_wmma_f32_16x16x32_f16(
                        false, pfrag[u][f], false, bf, (short)0, Ofrag[u][t], false, false);
            }
        }
    }

    // ---- epilogue: write UNNORMALIZED partials for this split ----
    float* Op = Opart + (size_t)split * N * DIM;
    #pragma unroll
    for (int u = 0; u < MT; ++u)
        #pragma unroll
        for (int v = 0; v < 8; ++v) {
            const int row = rowBase + u * 16 + v + 8 * half;
            #pragma unroll
            for (int t = 0; t < 8; ++t)
                Op[(size_t)row * DIM + t * 16 + l16] = Ofrag[u][t][v];
            if (l16 == 0) {
                mpart[(size_t)split * N + row] = mrow[u][v];
                lpart[(size_t)split * N + row] = lrow[u][v];
            }
        }
}

// ---------------- split combine: logsumexp merge + normalize ----------------

__global__ void combine_kernel(const float* __restrict__ Opart,
                               const float* __restrict__ mpart,
                               const float* __restrict__ lpart,
                               float* __restrict__ out,
                               float* __restrict__ lse, int N) {
    int i = blockIdx.x * blockDim.x + threadIdx.x;   // over N*DIM
    if (i >= N * DIM) return;
    const int row = i / DIM;
    const int d   = i - row * DIM;

    float ms[SPLITS], ls[SPLITS];
    float mstar = -INFINITY;
    #pragma unroll
    for (int s = 0; s < SPLITS; ++s) {
        ms[s] = mpart[(size_t)s * N + row];
        ls[s] = lpart[(size_t)s * N + row];
        mstar = fmaxf(mstar, ms[s]);
    }
    float lsum = 0.f, o = 0.f;
    #pragma unroll
    for (int s = 0; s < SPLITS; ++s) {
        float w = __expf(ms[s] - mstar);
        lsum += ls[s] * w;
        o += Opart[((size_t)s * N + row) * DIM + d] * w;
    }
    out[i] = o / lsum;
    if (d == 0) lse[row] = mstar + __logf(lsum);
}

// ---------------- host launch ----------------

extern "C" void kernel_launch(void* const* d_in, const int* in_sizes, int n_in,
                              void* d_out, int out_size, void* d_ws, size_t ws_size,
                              hipStream_t stream) {
    const float* Q   = (const float*)d_in[0];
    const float* K   = (const float*)d_in[1];
    const float* V   = (const float*)d_in[2];
    const float* h   = (const float*)d_in[3];
    const float* ldq = (const float*)d_in[4];
    const float* ldk = (const float*)d_in[5];

    const int N = in_sizes[3];   // h has N entries
    const int M = in_sizes[5];   // log_d_k has M entries

    char* ws = (char*)d_ws;
    size_t o = 0;
    _Float16* Qh = (_Float16*)(ws + o); o += (size_t)N * DIM * sizeof(_Float16);
    _Float16* Kh = (_Float16*)(ws + o); o += (size_t)M * DIM * sizeof(_Float16);
    _Float16* Vt = (_Float16*)(ws + o); o += (size_t)M * DIM * sizeof(_Float16);
    float* abias = (float*)(ws + o);    o += (size_t)N * sizeof(float);
    float* bbias = (float*)(ws + o);    o += (size_t)M * sizeof(float);
    float* Opart = (float*)(ws + o);    o += (size_t)SPLITS * N * DIM * sizeof(float);
    float* mpart = (float*)(ws + o);    o += (size_t)SPLITS * N * sizeof(float);
    float* lpart = (float*)(ws + o);    o += (size_t)SPLITS * N * sizeof(float);

    float* out = (float*)d_out;
    float* lse = out + (size_t)N * DIM;

    const int nq = N * DIM;
    const int nk = M * DIM;
    cvt_f16_kernel    <<<(nq + 255) / 256, 256, 0, stream>>>(Q, Qh, nq);
    cvt_f16_kernel    <<<(nk + 255) / 256, 256, 0, stream>>>(K, Kh, nk);
    transpose_v_kernel<<<(nk + 255) / 256, 256, 0, stream>>>(V, Vt, M);
    row_bias_kernel   <<<(N + 255) / 256, 256, 0, stream>>>(Q, ldq, h, abias, N);
    col_bias_kernel   <<<(M + 255) / 256, 256, 0, stream>>>(K, ldk, h, bbias, M);

    dim3 grid(N / BM, SPLITS);
    flash_doob_kernel<<<grid, BLOCK, 0, stream>>>(Qh, Kh, Vt, abias, bbias,
                                                  Opart, mpart, lpart, N, M);

    combine_kernel<<<(N * DIM + 255) / 256, 256, 0, stream>>>(Opart, mpart, lpart,
                                                              out, lse, N);
}
